// Encoder_22608707846323
// MI455X (gfx1250) — matrix operand order repaired and checked by reading, not compiled, not run on hardware
//
#include <hip/hip_runtime.h>
#include <hip/hip_bf16.h>
#include <stdint.h>

#define N_NODES 100000
#define N_EDGES 3200000
#define F 128
#define LDR 136   // padded LDS row stride in shorts (68 dwords -> bank stride 4, no 16-way conflicts)

typedef __attribute__((ext_vector_type(16))) __bf16 v16bf;
typedef __attribute__((ext_vector_type(8)))  float  v8f;

union Frag16 { v16bf v; unsigned short u[16]; };

__device__ __forceinline__ unsigned short f32_to_bf16_bits(float f) {
    union { float f; unsigned u; } x; x.f = f;
    // round-to-nearest-even
    unsigned r = x.u + 0x7FFFu + ((x.u >> 16) & 1u);
    return (unsigned short)(r >> 16);
}

// ---------------------------------------------------------------------------
// Kernel 1: degree accumulation (float atomics, arrays zeroed beforehand)
// ---------------------------------------------------------------------------
__global__ __launch_bounds__(256) void degree_kernel(
    const long long* __restrict__ src, const long long* __restrict__ dst,
    float* __restrict__ deg_out, float* __restrict__ deg_in) {
    int stride = gridDim.x * blockDim.x;
    for (int e = blockIdx.x * blockDim.x + threadIdx.x; e < N_EDGES; e += stride) {
        atomicAdd(&deg_out[(int)src[e]], 1.0f);
        atomicAdd(&deg_in [(int)dst[e]], 1.0f);
    }
}

// ---------------------------------------------------------------------------
// Kernel 2: deg -> 1/sqrt(max(deg,1)) in place, for both arrays
// ---------------------------------------------------------------------------
__global__ __launch_bounds__(256) void rsqrt_kernel(
    float* __restrict__ deg_out, float* __restrict__ deg_in) {
    int stride = gridDim.x * blockDim.x;
    for (int i = blockIdx.x * blockDim.x + threadIdx.x; i < N_NODES; i += stride) {
        deg_out[i] = rsqrtf(fmaxf(deg_out[i], 1.0f));
        deg_in [i] = rsqrtf(fmaxf(deg_in [i], 1.0f));
    }
}

// ---------------------------------------------------------------------------
// Kernel 3: SpMM with folded source normalization.
//   m[dst] += x[src] * dout_is[src]
// One wave (32 lanes) per edge; lane handles one float4 (128 floats/row).
// Gather is a coalesced 512B burst; x and m are L2-resident (51.2MB each in
// the 192MB L2), so the f32 atomics resolve at the L2 atomic units.
// ---------------------------------------------------------------------------
__global__ __launch_bounds__(256) void spmm_kernel(
    const float* __restrict__ x, const long long* __restrict__ src,
    const long long* __restrict__ dst, const float* __restrict__ dout_is,
    float* __restrict__ m) {
    int wave   = (blockIdx.x * blockDim.x + threadIdx.x) >> 5;
    int lane   = threadIdx.x & 31;
    int nwaves = (gridDim.x * blockDim.x) >> 5;
    for (int e = wave; e < N_EDGES; e += nwaves) {
        int s = (int)src[e];
        int d = (int)dst[e];
        float sc = dout_is[s];
        float4 v = ((const float4*)(x + (size_t)s * F))[lane];
        float* mo = m + (size_t)d * F + lane * 4;
        atomicAdd(mo + 0, v.x * sc);
        atomicAdd(mo + 1, v.y * sc);
        atomicAdd(mo + 2, v.z * sc);
        atomicAdd(mo + 3, v.w * sc);
    }
}

// ---------------------------------------------------------------------------
// Kernel 4: fused (din_is scale) + GEMM (m @ W) + bias + PReLU via WMMA bf16.
// Block = 256 threads = 8 waves; block computes a 16-row x 128-col tile;
// wave w owns N-tile [16w, 16w+16). K=128 -> 4 x v_wmma_f32_16x16x32_bf16.
// A tile staged K-major; W staged TRANSPOSED (column-major, sWt[n][k]) so
// every fragment load is a contiguous run -> ds_load_b128, no scalar u16
// gathers. Rows padded to LDR=136 shorts to spread LDS banks.
// ---------------------------------------------------------------------------
__global__ __launch_bounds__(256) void gemm_wmma_prelu_kernel(
    const float* __restrict__ A,       // m: [N_NODES, 128] aggregated messages
    const float* __restrict__ din_is,  // [N_NODES]
    const float* __restrict__ W,       // [128, 128] row-major (k, n)
    const float* __restrict__ bias,    // [128]
    const float* __restrict__ pa,      // [128] PReLU slope
    float* __restrict__ out) {         // [N_NODES, 128]
    __shared__ unsigned short sA [16 * LDR];   // ~4.3 KB  bf16 A tile (row r, K-major)
    __shared__ unsigned short sWt[F * LDR];    // ~34.8 KB bf16 W^T (col n, K-major)

    const int tid  = threadIdx.x;
    const int row0 = blockIdx.x * 16;          // 100000 % 16 == 0 -> no tail

    // Stage W transposed. Global read coalesced; one-time strided LDS write.
    for (int i = tid; i < F * F; i += 256) {
        int k = i >> 7, n = i & (F - 1);
        sWt[n * LDR + k] = f32_to_bf16_bits(W[i]);
    }
    // Stage A tile with folded dest-normalization.
    for (int i = tid; i < 16 * F; i += 256) {
        int r = i >> 7, k = i & (F - 1);
        int row = row0 + r;
        sA[r * LDR + k] = f32_to_bf16_bits(A[(size_t)row * F + k] * din_is[row]);
    }
    __syncthreads();

    const int wave = tid >> 5;
    const int lane = tid & 31;
    const int n0   = wave * 16;
    const int nl   = lane & 15;   // row within M-tile (A) / column within N-tile (B)
    const bool hi  = lane >= 16;

    v8f acc = {};
    #pragma unroll
    for (int s = 0; s < 4; ++s) {
        const int kb = s * 32;
        Frag16 a, b;
        // A 16x32 bf16 layout (ISA 7.12.2): lanes<16 -> elems 0..7 = K kb..kb+7,
        // elems 8..15 = K kb+16..; lanes>=16 -> kb+8.. and kb+24..
        const int kA0 = kb + (hi ? 8 : 0);
        const int kA1 = kb + (hi ? 24 : 16);
        #pragma unroll
        for (int e = 0; e < 8; ++e) {
            a.u[e]     = sA[nl * LDR + kA0 + e];   // contiguous 16B -> ds_load_b128
            a.u[e + 8] = sA[nl * LDR + kA1 + e];   // contiguous 16B -> ds_load_b128
        }
        // B 32x16 bf16: same K order as A: lanes<16 -> elems 0..7 = K kb..kb+7,
        // elems 8..15 = K kb+16..; lanes>=16 -> kb+8.. and kb+24.. -> 2x ds_load_b128.
        const int kB0 = kb + (hi ? 8 : 0);
        const int kB1 = kb + (hi ? 24 : 16);
        #pragma unroll
        for (int e = 0; e < 8; ++e) {
            b.u[e]     = sWt[(n0 + nl) * LDR + kB0 + e];
            b.u[e + 8] = sWt[(n0 + nl) * LDR + kB1 + e];
        }

        acc = __builtin_amdgcn_wmma_f32_16x16x32_bf16(
            /*neg_a=*/false, a.v, /*neg_b=*/false, b.v,
            /*c_mod=*/(short)0, acc, /*reuse_a=*/false, /*reuse_b=*/false);
    }

    // Epilogue: bias + PReLU, write f32.
    // D layout: VGPR r -> row r (lanes 0..15) / row r+8 (lanes 16..31), col = n0+nl.
    const int col = n0 + nl;
    const float bi = bias[col];
    const float al = pa[col];
    #pragma unroll
    for (int r = 0; r < 8; ++r) {
        int row = row0 + r + (hi ? 8 : 0);
        float v = acc[r] + bi;
        v = (v >= 0.0f) ? v : al * v;
        out[(size_t)row * F + col] = v;
    }
}

// ---------------------------------------------------------------------------
// Host-side launch sequence (graph-capture safe: only async stream ops).
// ---------------------------------------------------------------------------
extern "C" void kernel_launch(void* const* d_in, const int* in_sizes, int n_in,
                              void* d_out, int out_size, void* d_ws, size_t ws_size,
                              hipStream_t stream) {
    (void)in_sizes; (void)n_in; (void)out_size; (void)ws_size;
    const float*     feat = (const float*)d_in[0];
    const long long* ei   = (const long long*)d_in[1];  // int64 [2, E]
    const float*     W1   = (const float*)d_in[2];
    const float*     b1   = (const float*)d_in[3];
    const float*     W2   = (const float*)d_in[4];
    const float*     b2   = (const float*)d_in[5];
    const float*     pa   = (const float*)d_in[6];
    float*           out  = (float*)d_out;

    const long long* src = ei;
    const long long* dst = ei + N_EDGES;

    // Workspace layout: [dout_is | din_is | m]
    float* dout_is = (float*)d_ws;
    float* din_is  = dout_is + N_NODES;
    float* m       = din_is + N_NODES;
    const size_t mBytes = (size_t)N_NODES * F * sizeof(float);  // 51.2 MB

    // Zero degrees + aggregation buffer (stream memset is capture-legal).
    hipMemsetAsync(dout_is, 0, 2 * (size_t)N_NODES * sizeof(float), stream);
    hipMemsetAsync(m, 0, mBytes, stream);

    // Degrees -> inverse sqrt norms.
    degree_kernel<<<2048, 256, 0, stream>>>(src, dst, dout_is, din_is);
    rsqrt_kernel<<<(N_NODES + 255) / 256, 256, 0, stream>>>(dout_is, din_is);

    // ---- Layer 1 ----
    spmm_kernel<<<8192, 256, 0, stream>>>(feat, src, dst, dout_is, m);
    gemm_wmma_prelu_kernel<<<N_NODES / 16, 256, 0, stream>>>(m, din_is, W1, b1, pa, out);

    // ---- Layer 2 ----
    hipMemsetAsync(m, 0, mBytes, stream);
    spmm_kernel<<<8192, 256, 0, stream>>>(out, src, dst, dout_is, m);
    gemm_wmma_prelu_kernel<<<N_NODES / 16, 256, 0, stream>>>(m, din_is, W2, b2, pa, out);
}
